// LlamaAttention_63711544869302
// MI455X (gfx1250) — matrix operand order repaired and checked by reading, not compiled, not run on hardware
//
#include <hip/hip_runtime.h>
#include <hip/hip_bf16.h>
#include <math.h>
#include <stdint.h>

typedef __attribute__((ext_vector_type(16))) _Float16 v16h;
typedef __attribute__((ext_vector_type(8)))  _Float16 v8h;
typedef __attribute__((ext_vector_type(8)))  float    v8f;

#define B_   2
#define L_   2048
#define D_   2048
#define H_   32
#define HKV_ 8
#define HD_  64

// ---- CDNA5 staging path selection (TDM > async-to-LDS > plain) ----
#if defined(__has_builtin)
#  if __has_builtin(__builtin_amdgcn_tensor_load_to_lds) && __has_builtin(__builtin_amdgcn_s_wait_tensorcnt)
#    define STAGE_MODE 2
#  elif __has_builtin(__builtin_amdgcn_global_load_async_to_lds_b128)
#    define STAGE_MODE 1
#  else
#    define STAGE_MODE 0
#  endif
#else
#  define STAGE_MODE 0
#endif

__device__ __forceinline__ v8f wmma_f16(v16h a, v16h b, v8f c) {
  // D = A(16x32 f16) * B(32x16 f16) + C(16x16 f32)
  return __builtin_amdgcn_wmma_f32_16x16x32_f16(false, a, false, b, (short)0, c, false, false);
}

#if STAGE_MODE == 2
typedef __attribute__((ext_vector_type(4))) unsigned int u32x4;
typedef __attribute__((ext_vector_type(8))) int          i32x8;
typedef __attribute__((ext_vector_type(4))) int          i32x4;

// TDM 2D tile load: tile_k (contiguous, dim0) x rows (dim1), row stride = k_stride elems,
// f16 elements (data_size code 1). D# per CDNA5 ISA ch.8 (groups 2/3 zero => 2D tensor).
__device__ __forceinline__ void tdm_load_2d(const _Float16* gsrc, unsigned lds_byte_off,
                                            int rows, int k_stride, int tile_k, int tot_rows) {
  unsigned long long ga = (unsigned long long)(uintptr_t)gsrc;
  unsigned long long g0lo = 1ull /*count=1*/ | ((unsigned long long)lds_byte_off << 32);
  unsigned long long g0hi = ga /*57-bit global addr at [120:64]*/ | (2ull << 62) /*type=image*/;
  u32x4 g0 = { (unsigned)g0lo, (unsigned)(g0lo >> 32), (unsigned)g0hi, (unsigned)(g0hi >> 32) };

  unsigned long long q0 = 0x10000ull /*data_size=1(2B), wg_mask=0*/
                        | ((unsigned long long)((unsigned)k_stride & 0xFFFFu) << 48);
  unsigned long long q1 = (unsigned long long)((unsigned)k_stride >> 16)
                        | ((unsigned long long)(unsigned)tot_rows << 16)
                        | ((unsigned long long)(unsigned)tile_k << 48);
  unsigned long long q2 = (unsigned long long)(unsigned)rows
                        | ((unsigned long long)(unsigned)k_stride << 32);
  unsigned long long q3 = 0;
  i32x8 g1 = { (int)q0, (int)(q0 >> 32), (int)q1, (int)(q1 >> 32),
               (int)q2, (int)(q2 >> 32), (int)q3, (int)(q3 >> 32) };
  i32x4 gz = { 0, 0, 0, 0 };
#if __clang_major__ >= 23
  i32x8 gz8 = { 0, 0, 0, 0, 0, 0, 0, 0 };
  __builtin_amdgcn_tensor_load_to_lds(g0, g1, gz, gz, gz8, 0);
#else
  __builtin_amdgcn_tensor_load_to_lds(g0, g1, gz, gz, 0);
#endif
}
#endif

// wave-level staging wait: stage n landed, `inflight` newer ops still outstanding
__device__ __forceinline__ void stage_wait(bool more) {
#if STAGE_MODE == 2
  if (more) __builtin_amdgcn_s_wait_tensorcnt(2);
  else      __builtin_amdgcn_s_wait_tensorcnt(0);
#elif STAGE_MODE == 1
#if __has_builtin(__builtin_amdgcn_s_wait_asynccnt)
  if (more) __builtin_amdgcn_s_wait_asynccnt(4);
  else      __builtin_amdgcn_s_wait_asynccnt(0);
#else
  if (more) asm volatile("s_wait_asynccnt 4" ::: "memory");
  else      asm volatile("s_wait_asynccnt 0" ::: "memory");
#endif
#else
  (void)more;
#endif
}

// ---------------- f32 -> f16 straight convert ----------------
__global__ void cvt_kernel(const float* __restrict__ src, _Float16* __restrict__ dst, int n) {
  for (int i = blockIdx.x * blockDim.x + threadIdx.x; i < n; i += gridDim.x * blockDim.x)
    dst[i] = (_Float16)src[i];
}

// ---------------- f32 [K][N] -> f16 transposed [N][K] ----------------
__global__ void cvtT_kernel(const float* __restrict__ w, _Float16* __restrict__ wt, int K, int N) {
  int total = K * N;
  for (int i = blockIdx.x * blockDim.x + threadIdx.x; i < total; i += gridDim.x * blockDim.x) {
    int n = i / K;
    int k = i - n * K;
    wt[i] = (_Float16)w[(size_t)k * N + n];
  }
}

// ---------------- LDS-staged WMMA GEMM ----------------
// C[M,N] = A[M,K](f16) * BT[N,K](f16). Block tile 64(M) x 128(N), K-step 32, double-buffered LDS.
template <int MODE>
__global__ __launch_bounds__(256) void gemm_kernel(const _Float16* __restrict__ A,
                                                   const _Float16* __restrict__ BT,
                                                   void* __restrict__ outp,
                                                   int M, int N, int K, int Hn) {
  __shared__ _Float16 smem[2 * 6144];  // per stage: A 64x32 (4KB) + B 128x32 (8KB)
  const int tid  = threadIdx.x;
  const int wave = tid >> 5;
  const int lane = tid & 31;
  const int ml   = lane & 15;
  const int hi   = (lane >> 4) & 1;
  const int wave_m = wave >> 2;
  const int wave_n = wave & 3;
  const int tile_m = blockIdx.y * 64;
  const int tile_n = blockIdx.x * 128;
  const int aoff = hi ? 16 : 0;
  const int boff = hi ? 16 : 0;

  auto issue_stage = [&](int buf, int k0) {
    unsigned stage_byte = (unsigned)buf * 12288u;
#if STAGE_MODE == 2
    if (wave == 0) {
      tdm_load_2d(A  + (size_t)tile_m * K + k0, stage_byte,          64,  K, 32, M);
      tdm_load_2d(BT + (size_t)tile_n * K + k0, stage_byte + 4096u, 128,  K, 32, N);
    }
#else
#pragma unroll
    for (int t = 0; t < 3; ++t) {
      unsigned off = (unsigned)(tid + t * 256) * 16u;
      const _Float16* g;
      if (off < 4096u) {
        g = A + (size_t)(tile_m + (off >> 6)) * K + k0 + ((off & 63u) >> 1);
      } else {
        unsigned ob = off - 4096u;
        g = BT + (size_t)(tile_n + (ob >> 6)) * K + k0 + ((ob & 63u) >> 1);
      }
#if STAGE_MODE == 1
      __builtin_amdgcn_global_load_async_to_lds_b128(
          (__attribute__((address_space(1))) const void*)(uintptr_t)g,
          (__attribute__((address_space(3))) void*)(uintptr_t)(stage_byte + off), 0, 0);
#else
      *(v8h*)(smem + (size_t)buf * 6144 + (off >> 1)) = *(const v8h*)g;
#endif
    }
#endif
  };

  v8f acc00 = {}, acc01 = {}, acc10 = {}, acc11 = {};
  issue_stage(0, 0);
  int cur = 0;
  for (int k0 = 0; k0 < K; k0 += 32, cur ^= 1) {
    bool more = (k0 + 32) < K;
    if (more) issue_stage(cur ^ 1, k0 + 32);
    stage_wait(more);
    __syncthreads();

    const _Float16* As = smem + cur * 6144;
    const _Float16* Bs = As + 2048;
    v16h a0, a1, b0, b1;
    {
      const _Float16* ap0 = As + (wave_m * 32 + ml) * 32 + aoff;
      const _Float16* ap1 = As + (wave_m * 32 + 16 + ml) * 32 + aoff;
      v8h l0 = *(const v8h*)ap0, h0 = *(const v8h*)(ap0 + 8);
      v8h l1 = *(const v8h*)ap1, h1 = *(const v8h*)(ap1 + 8);
#pragma unroll
      for (int j = 0; j < 8; ++j) {
        a0[j] = l0[j]; a0[8 + j] = h0[j];
        a1[j] = l1[j]; a1[8 + j] = h1[j];
      }
      b0 = *(const v16h*)(Bs + (wave_n * 32 + ml) * 32 + boff);
      b1 = *(const v16h*)(Bs + (wave_n * 32 + 16 + ml) * 32 + boff);
    }
    acc00 = wmma_f16(a0, b0, acc00);
    acc01 = wmma_f16(a0, b1, acc01);
    acc10 = wmma_f16(a1, b0, acc10);
    acc11 = wmma_f16(a1, b1, acc11);
    __syncthreads();
  }

  const v8f* accs[4] = { &acc00, &acc01, &acc10, &acc11 };
#pragma unroll
  for (int s = 0; s < 4; ++s) {
    int sm = s >> 1, sn = s & 1;
    const v8f& a = *accs[s];
#pragma unroll
    for (int r = 0; r < 8; ++r) {
      int row = tile_m + wave_m * 32 + sm * 16 + r + hi * 8;
      int col = tile_n + wave_n * 32 + sn * 16 + ml;
      if constexpr (MODE == 1) {
        ((float*)outp)[(size_t)row * N + col] = a[r];
      } else {
        int bb = row >> 11, l = row & 2047;
        int h  = col >> 6,  hd = col & 63;
        _Float16* o = (_Float16*)outp;
        if constexpr (MODE == 0)
          o[(((size_t)bb * Hn + h) * L_ + l) * HD_ + hd] = (_Float16)a[r];
        else
          o[(((size_t)bb * Hn + h) * HD_ + hd) * L_ + l] = (_Float16)a[r];
      }
    }
  }
}

// ---------------- in-place f16 RoPE on [BH][L][HD] ----------------
__global__ void rope_kernel(_Float16* __restrict__ t, const float* __restrict__ cs,
                            const float* __restrict__ sn, int total) {
  for (int id = blockIdx.x * blockDim.x + threadIdx.x; id < total; id += gridDim.x * blockDim.x) {
    int i  = id & 31;
    int l  = (id >> 5) & 2047;
    int bh = id >> 16;
    size_t base = ((size_t)bh * L_ + l) * HD_ + 2 * i;
    float tr = (float)t[base], ti = (float)t[base + 1];
    float c = cs[l * 32 + i], s = sn[l * 32 + i];
    t[base]     = (_Float16)(tr * c - ti * s);
    t[base + 1] = (_Float16)(tr * s + ti * c);
  }
}

// ---------------- causal flash attention ----------------
// 4 waves/block, one 16-row Q tile each, same (b, kv-head): K/V tiles staged to LDS once per
// block (TDM/async double-buffer), clean/masked K-block paths split on a wave-uniform predicate.
// Q: [B][H][L][HD], Kc: [B][HKV][L][HD], Vt: [B][HKV][HD][L], outA: [B][L][H*HD]
__global__ __launch_bounds__(128) void flash_kernel(const _Float16* __restrict__ Q,
                                                    const _Float16* __restrict__ Kc,
                                                    const _Float16* __restrict__ Vt,
                                                    _Float16* __restrict__ outA) {
  __shared__ _Float16 sm[10240];  // [0,8192): 2 stages x (K 32x64 + V 64x32); [8192,10240): P 4x512
  const int tid  = threadIdx.x;
  const int wave = tid >> 5;
  const int lane = tid & 31;
  const int ml   = lane & 15;
  const int hi   = (lane >> 4) & 1;
  const int qbase = blockIdx.x * 64 + wave * 16;
  const int bmax  = blockIdx.x * 64 + 63;        // last causal row in this block
  const int h = blockIdx.y, b = blockIdx.z;
  const int kvh = h >> 2;

  const _Float16* qrow = Q + (((size_t)(b * H_ + h)) * L_ + (qbase + ml)) * HD_;
  v16h aq[2];
#pragma unroll
  for (int c = 0; c < 2; ++c) {
    int kb = c * 32 + (hi ? 16 : 0);
    v8h lo8 = *(const v8h*)(qrow + kb);
    v8h hi8 = *(const v8h*)(qrow + kb + 8);
#pragma unroll
    for (int j = 0; j < 8; ++j) { aq[c][j] = lo8[j]; aq[c][8 + j] = hi8[j]; }
  }

  const _Float16* kbp = Kc + ((size_t)(b * HKV_ + kvh)) * L_ * HD_;
  const _Float16* vbp = Vt + ((size_t)(b * HKV_ + kvh)) * HD_ * L_;
  _Float16* myP = sm + 8192 + wave * 512;

  auto issue = [&](int buf, int k0) {
    unsigned sb = (unsigned)buf * 8192u;
#if STAGE_MODE == 2
    if (wave == 0) {
      tdm_load_2d(kbp + (size_t)k0 * HD_, sb,          32, HD_, 64, L_);  // K: [32 kpos][64 hd]
      tdm_load_2d(vbp + k0,               sb + 4096u,  64, L_,  32, HD_); // V: [64 hd][32 kpos]
    }
#else
#pragma unroll
    for (int t = 0; t < 4; ++t) {                       // 512 x 16B chunks / 128 threads
      unsigned off = (unsigned)(tid + t * 128) * 16u;
      const _Float16* g;
      if (off < 4096u) g = kbp + (size_t)k0 * HD_ + (off >> 1);
      else { unsigned ob = off - 4096u;
             g = vbp + (size_t)(ob >> 6) * L_ + k0 + ((ob & 63u) >> 1); }
#if STAGE_MODE == 1
      __builtin_amdgcn_global_load_async_to_lds_b128(
          (__attribute__((address_space(1))) const void*)(uintptr_t)g,
          (__attribute__((address_space(3))) void*)(uintptr_t)(sb + off), 0, 0);
#else
      *(v8h*)(sm + (size_t)buf * 4096 + (off >> 1)) = *(const v8h*)g;
#endif
    }
#endif
  };

  float mrow[8], lrow[8];
  v8f o0 = {}, o1 = {}, o2 = {}, o3 = {};
#pragma unroll
  for (int r = 0; r < 8; ++r) { mrow[r] = -3.0e38f; lrow[r] = 0.0f; }

  issue(0, 0);
  int cur = 0;
  for (int k0 = 0; k0 <= bmax; k0 += 32, cur ^= 1) {
    bool more = (k0 + 32) <= bmax;
    if (more) issue(cur ^ 1, k0 + 32);
    stage_wait(more);
    __syncthreads();

    if (k0 <= qbase + 15) {          // wave-uniform: this wave still in causal range
      const _Float16* sK = sm + cur * 4096;
      const _Float16* sV = sK + 2048;
      v8f s[2];
#pragma unroll
      for (int t = 0; t < 2; ++t) {
        const _Float16* krow = sK + (size_t)(t * 16 + ml) * 64;
        v16h bk0 = *(const v16h*)(krow + (hi ? 16 : 0));
        v16h bk1 = *(const v16h*)(krow + 32 + (hi ? 16 : 0));
        v8f acc = {};
        acc  = wmma_f16(aq[0], bk0, acc);
        s[t] = wmma_f16(aq[1], bk1, acc);
      }

      const bool need_mask = (k0 + 31) > qbase;  // wave-uniform: touches causal diagonal
      float p0a[8], p1a[8];
#pragma unroll
      for (int r = 0; r < 8; ++r) {
        int rrow = qbase + r + hi * 8;
        float v0 = s[0][r] * 0.125f;
        float v1 = s[1][r] * 0.125f;
        if (need_mask) {
          if (k0 + ml > rrow)      v0 = -1.0e30f;
          if (k0 + 16 + ml > rrow) v1 = -1.0e30f;
        }
        float bm = fmaxf(v0, v1);
#pragma unroll
        for (int msk = 1; msk <= 8; msk <<= 1) bm = fmaxf(bm, __shfl_xor(bm, msk, 32));
        float mnew  = fmaxf(mrow[r], bm);
        float alpha = __expf(mrow[r] - mnew);
        mrow[r] = mnew;
        float p0 = __expf(v0 - mnew);
        float p1 = __expf(v1 - mnew);
        float ps = p0 + p1;
#pragma unroll
        for (int msk = 1; msk <= 8; msk <<= 1) ps += __shfl_xor(ps, msk, 32);
        lrow[r] = lrow[r] * alpha + ps;
        o0[r] *= alpha; o1[r] *= alpha; o2[r] *= alpha; o3[r] *= alpha;
        p0a[r] = p0; p1a[r] = p1;
      }

#pragma unroll
      for (int r = 0; r < 8; ++r) {
        int rr = r + hi * 8;
        myP[rr * 32 + ml]      = (_Float16)p0a[r];
        myP[rr * 32 + 16 + ml] = (_Float16)p1a[r];
      }
      v16h pa;
      {
        int kb = hi ? 16 : 0;
#pragma unroll
        for (int j = 0; j < 8; ++j) {
          pa[j]     = myP[ml * 32 + kb + j];
          pa[8 + j] = myP[ml * 32 + kb + 8 + j];
        }
      }
      {
        v16h bv0 = *(const v16h*)(sV + (size_t)( 0 + ml) * 32 + (hi ? 16 : 0));
        v16h bv1 = *(const v16h*)(sV + (size_t)(16 + ml) * 32 + (hi ? 16 : 0));
        v16h bv2 = *(const v16h*)(sV + (size_t)(32 + ml) * 32 + (hi ? 16 : 0));
        v16h bv3 = *(const v16h*)(sV + (size_t)(48 + ml) * 32 + (hi ? 16 : 0));
        o0 = wmma_f16(pa, bv0, o0);
        o1 = wmma_f16(pa, bv1, o1);
        o2 = wmma_f16(pa, bv2, o2);
        o3 = wmma_f16(pa, bv3, o3);
      }
    }
    __syncthreads();   // protect stage buffer before refill
  }

#pragma unroll
  for (int r = 0; r < 8; ++r) {
    int rrow = qbase + r + hi * 8;
    float inv = 1.0f / lrow[r];
    _Float16* orow = outA + ((size_t)b * L_ + rrow) * (H_ * HD_) + h * HD_;
    orow[ 0 + ml] = (_Float16)(o0[r] * inv);
    orow[16 + ml] = (_Float16)(o1[r] * inv);
    orow[32 + ml] = (_Float16)(o2[r] * inv);
    orow[48 + ml] = (_Float16)(o3[r] * inv);
  }
}

extern "C" void kernel_launch(void* const* d_in, const int* in_sizes, int n_in,
                              void* d_out, int out_size, void* d_ws, size_t ws_size,
                              hipStream_t stream) {
  (void)in_sizes; (void)n_in; (void)out_size; (void)ws_size;
  const float* x  = (const float*)d_in[0];
  const float* wq = (const float*)d_in[1];
  const float* wk = (const float*)d_in[2];
  const float* wv = (const float*)d_in[3];
  const float* wo = (const float*)d_in[4];
  const float* fc = (const float*)d_in[5];
  const float* fs = (const float*)d_in[6];
  // d_in[7] = additive causal mask: applied analytically inside flash_kernel.

  char* ws = (char*)d_ws;
  size_t off = 0;
  auto carve = [&](size_t bytes) { char* p = ws + off; off += bytes; return p; };
  _Float16* xh  = (_Float16*)carve((size_t)B_ * L_ * D_ * 2);
  _Float16* wqT = (_Float16*)carve((size_t)D_ * H_ * HD_ * 2);
  _Float16* wkT = (_Float16*)carve((size_t)D_ * HKV_ * HD_ * 2);
  _Float16* wvT = (_Float16*)carve((size_t)D_ * HKV_ * HD_ * 2);
  _Float16* woT = (_Float16*)carve((size_t)D_ * D_ * 2);
  _Float16* qb  = (_Float16*)carve((size_t)B_ * H_ * L_ * HD_ * 2);
  _Float16* kq  = (_Float16*)carve((size_t)B_ * HKV_ * L_ * HD_ * 2);
  _Float16* vT  = (_Float16*)carve((size_t)B_ * HKV_ * HD_ * L_ * 2);
  _Float16* at  = (_Float16*)carve((size_t)B_ * L_ * H_ * HD_ * 2);

  cvt_kernel <<<2048, 256, 0, stream>>>(x,  xh,  B_ * L_ * D_);
  cvtT_kernel<<<2048, 256, 0, stream>>>(wq, wqT, D_, H_ * HD_);
  cvtT_kernel<<<1024, 256, 0, stream>>>(wk, wkT, D_, HKV_ * HD_);
  cvtT_kernel<<<1024, 256, 0, stream>>>(wv, wvT, D_, HKV_ * HD_);
  cvtT_kernel<<<2048, 256, 0, stream>>>(wo, woT, D_, D_);

  dim3 blk(256);
  gemm_kernel<0><<<dim3((H_ * HD_) / 128,   (B_ * L_) / 64), blk, 0, stream>>>(xh, wqT, qb, B_ * L_, H_ * HD_,   D_, H_);
  gemm_kernel<0><<<dim3((HKV_ * HD_) / 128, (B_ * L_) / 64), blk, 0, stream>>>(xh, wkT, kq, B_ * L_, HKV_ * HD_, D_, HKV_);
  gemm_kernel<2><<<dim3((HKV_ * HD_) / 128, (B_ * L_) / 64), blk, 0, stream>>>(xh, wvT, vT, B_ * L_, HKV_ * HD_, D_, HKV_);

  rope_kernel<<<4096, 256, 0, stream>>>(qb, fc, fs, B_ * H_ * L_ * 32);
  rope_kernel<<<1024, 256, 0, stream>>>(kq, fc, fs, B_ * HKV_ * L_ * 32);

  flash_kernel<<<dim3(L_ / 64, H_, B_), dim3(128), 0, stream>>>(qb, kq, vT, at);

  gemm_kernel<1><<<dim3(D_ / 128, (B_ * L_) / 64), blk, 0, stream>>>(at, woT, d_out, B_ * L_, D_, D_, 0);
}
